// MultiHeadAttention_81260781240694
// MI455X (gfx1250) — matrix operand order repaired and checked by reading, not compiled, not run on hardware
//
#include <hip/hip_runtime.h>
#include <hip/hip_bf16.h>

#define D_MODEL 1024
#define NUM_HEADS 16
#define DK 64
#define SEQ 2048
#define BATCH 2
#define ROWS (BATCH * SEQ) /* 4096 */

typedef __attribute__((ext_vector_type(16))) __bf16 v16bf;
typedef __attribute__((ext_vector_type(8)))  __bf16 bf16x8;
typedef __attribute__((ext_vector_type(8)))  float  v8f;
typedef __attribute__((ext_vector_type(8)))  unsigned short ushort8;

// f32 -> bf16 round-to-nearest-even
static __device__ __forceinline__ unsigned short f2bf(float f) {
  unsigned int u = __float_as_uint(f);
  u += 0x7FFFu + ((u >> 16) & 1u);
  return (unsigned short)(u >> 16);
}

// Build a 16-element bf16 WMMA fragment from two contiguous 8-element runs.
static __device__ __forceinline__ v16bf frag_from(const unsigned short* lo,
                                                  const unsigned short* hi) {
  bf16x8 l = __builtin_bit_cast(bf16x8, *(const ushort8*)lo);
  bf16x8 h = __builtin_bit_cast(bf16x8, *(const ushort8*)hi);
  return __builtin_shufflevector(l, h, 0,1,2,3,4,5,6,7,8,9,10,11,12,13,14,15);
}

// Generic (flat) LDS pointer -> 32-bit LDS byte address for async-to-LDS ops.
static __device__ __forceinline__ unsigned lds_addr(const void* p) {
  return (unsigned)(unsigned long long)(__attribute__((address_space(3))) const void*)p;
}

// CDNA5 async memory->LDS copy, 16 bytes per active lane. Tracked by ASYNCcnt.
static __device__ __forceinline__ void async_copy_b128(unsigned lds_off,
                                                       const unsigned short* gptr) {
  asm volatile("global_load_async_to_lds_b128 %0, %1, off"
               :: "v"(lds_off), "v"(gptr) : "memory");
}
static __device__ __forceinline__ void wait_async0() {
  asm volatile("s_wait_asynccnt 0x0" ::: "memory");
}

__global__ void cvt_f32_bf16(const float* __restrict__ src,
                             unsigned short* __restrict__ dst, int n) {
  int i = blockIdx.x * blockDim.x + threadIdx.x;
  if (i < n) dst[i] = f2bf(src[i]);
}

// ---------------------------------------------------------------------------
// Tiled bf16 WMMA GEMM: C[M,N] = A[M,K] @ B[K,N].
// Block tile 128x128x32, 256 threads = 8 waves (4M x 2N), each wave 32x64.
// Double-buffered LDS; A tile filled by async-to-LDS, B tile transposed
// through VGPRs (global loads hoisted ahead of the WMMA block for overlap).
// MODE 0: write f32 to outF.  MODE 1: scatter QKV (bf16), Q pre-scaled.
// ---------------------------------------------------------------------------
template <int MODE>
__launch_bounds__(256)
__global__ void gemm_bf16(const unsigned short* __restrict__ A,
                          const unsigned short* __restrict__ B,
                          int M, int N, int K,
                          float* __restrict__ outF,
                          unsigned short* __restrict__ qOut,
                          unsigned short* __restrict__ kOut,
                          unsigned short* __restrict__ vtOut) {
  constexpr int BM = 128, BN = 128, BK = 32, LDST = 40; // LDST mult of 8 -> 16B rows
  __shared__ __align__(16) unsigned short As[2][BM * LDST];
  __shared__ __align__(16) unsigned short Bt[2][BN * LDST]; // Bt[n][k]

  const int tid   = threadIdx.x;
  const int lane  = tid & 31;
  const int wid   = tid >> 5;
  const int waveM = wid >> 1; // 0..3
  const int waveN = wid & 1;  // 0..1
  const int blockM = blockIdx.y * BM;
  const int blockN = blockIdx.x * BN;
  const int ln16 = lane & 15;
  const int hi16 = lane >> 4; // 0 or 1

  v8f acc[2][4];
  const v8f vzero = {0.f, 0.f, 0.f, 0.f, 0.f, 0.f, 0.f, 0.f};
#pragma unroll
  for (int mi = 0; mi < 2; mi++)
#pragma unroll
    for (int ni = 0; ni < 4; ni++) acc[mi][ni] = vzero;

  const int alo = hi16 ? 8 : 0;   // A frag runs: k+{0,16} / k+{8,24}
  const int blo = hi16 ? 8 : 0;   // B frag runs: k+{0,16} / k+{8,24}

  auto issueA = [&](int kt, int buf) { // 128x32 tile, 2 async b128 per thread
#pragma unroll
    for (int r = 0; r < 2; r++) {
      int c = tid + r * 256;
      int row = c >> 2, cp = c & 3;
      async_copy_b128(lds_addr(&As[buf][row * LDST + cp * 8]),
                      &A[(size_t)(blockM + row) * K + kt + cp * 8]);
    }
  };
  auto loadB = [&](int kt, ushort8* breg) {
#pragma unroll
    for (int r = 0; r < 2; r++) {
      int c = tid + r * 256;
      int kk = c >> 4, n8 = c & 15;
      breg[r] = *(const ushort8*)&B[(size_t)(kt + kk) * N + blockN + n8 * 8];
    }
  };
  auto scatterB = [&](int buf, const ushort8* breg) {
#pragma unroll
    for (int r = 0; r < 2; r++) {
      int c = tid + r * 256;
      int kk = c >> 4, n8 = c & 15;
#pragma unroll
      for (int i = 0; i < 8; i++) Bt[buf][(n8 * 8 + i) * LDST + kk] = breg[r][i];
    }
  };

  // Prologue: fill buffer 0.
  ushort8 breg[2];
  issueA(0, 0);
  loadB(0, breg);
  scatterB(0, breg);
  wait_async0();
  __syncthreads();

  for (int kt = 0; kt < K; kt += BK) {
    const int cur = (kt >> 5) & 1, nxt = cur ^ 1;
    const bool hasnext = (kt + BK) < K;
    if (hasnext) {
      issueA(kt + BK, nxt);   // async, lands in other buffer
      loadB(kt + BK, breg);   // global->VGPR, overlaps WMMA below
    }

    v16bf afrag[2], bfrag[4];
#pragma unroll
    for (int mi = 0; mi < 2; mi++) {
      const unsigned short* p = &As[cur][(waveM * 32 + mi * 16 + ln16) * LDST];
      afrag[mi] = frag_from(p + alo, p + alo + 16);
    }
#pragma unroll
    for (int ni = 0; ni < 4; ni++) {
      const unsigned short* p = &Bt[cur][(waveN * 64 + ni * 16 + ln16) * LDST];
      bfrag[ni] = frag_from(p + blo, p + blo + 16);
    }
#pragma unroll
    for (int mi = 0; mi < 2; mi++)
#pragma unroll
      for (int ni = 0; ni < 4; ni++)
        acc[mi][ni] = __builtin_amdgcn_wmma_f32_16x16x32_bf16(
            false, afrag[mi], false, bfrag[ni], (short)0, acc[mi][ni], false, false);

    if (hasnext) scatterB(nxt, breg); // VGPR->LDS transpose scatter
    wait_async0();
    __syncthreads();
  }

  // Epilogue: C elem v <-> (row = v + 8*hi16, col = lane%16)
#pragma unroll
  for (int mi = 0; mi < 2; mi++) {
#pragma unroll
    for (int ni = 0; ni < 4; ni++) {
#pragma unroll
      for (int v = 0; v < 8; v++) {
        int row = blockM + waveM * 32 + mi * 16 + v + hi16 * 8;
        int col = blockN + waveN * 64 + ni * 16 + ln16;
        float val = acc[mi][ni][v];
        if (MODE == 0) {
          outF[(size_t)row * N + col] = val;
        } else {
          int b = row >> 11, s = row & (SEQ - 1);
          int which = col >> 10, hc = col & 1023;
          int h = hc >> 6, d = hc & (DK - 1);
          int bh = b * NUM_HEADS + h;
          if (which == 0)
            qOut[((size_t)bh * SEQ + s) * DK + d] = f2bf(val * 0.125f); // *dk^-0.5
          else if (which == 1)
            kOut[((size_t)bh * SEQ + s) * DK + d] = f2bf(val);
          else
            vtOut[((size_t)bh * DK + d) * SEQ + s] = f2bf(val); // V transposed
        }
      }
    }
  }
}

// ---------------------------------------------------------------------------
// Flash attention: block = 4 waves x 16 query rows = 64 q rows per (b,h).
// K / V^T tiles (64 keys) double-buffered in LDS via async-to-LDS copies;
// online softmax; S x S never materialized. Output bf16 as [b, s, h, d].
// ---------------------------------------------------------------------------
__launch_bounds__(128)
__global__ void flash_attn(const unsigned short* __restrict__ Q,
                           const unsigned short* __restrict__ Kk,
                           const unsigned short* __restrict__ Vt,
                           unsigned short* __restrict__ O) {
  __shared__ __align__(16) unsigned short Ks[2][64 * 64];  // K rows [t][d]
  __shared__ __align__(16) unsigned short Vs[2][64 * 64];  // V^T rows [d][t]
  __shared__ __align__(16) unsigned short Ps[4 * 16 * 72]; // per-wave P tiles

  const int tid = threadIdx.x, lane = tid & 31, wave = tid >> 5;
  const int ln16 = lane & 15, hi16 = lane >> 4;
  const int bh = blockIdx.y;
  const int qbase = blockIdx.x * 64;
  const unsigned short* Qbh = Q + (size_t)bh * SEQ * DK;
  const unsigned short* Kbh = Kk + (size_t)bh * SEQ * DK;
  const unsigned short* Vbh = Vt + (size_t)bh * DK * SEQ;

  const int alo = hi16 ? 8 : 0;
  const int blo = hi16 ? 8 : 0;

  auto issueKV = [&](int kt, int buf) { // 64x32 (cols 0..31) of each of the 2 64x64 tiles, 4 async b128/thread
#pragma unroll
    for (int r = 0; r < 2; r++) {
      int c = tid + r * 128;
      int rw = c >> 2, cp = c & 3;
      async_copy_b128(lds_addr(&Ks[buf][rw * 64 + cp * 8]),
                      &Kbh[(size_t)(kt + rw) * DK + cp * 8]);
      async_copy_b128(lds_addr(&Vs[buf][rw * 64 + cp * 8]),
                      &Vbh[(size_t)rw * SEQ + kt + cp * 8]);
    }
  };

  // Q fragments stay resident in VGPRs for the whole kernel.
  v16bf qf[2];
  {
    const unsigned short* qp = Qbh + (size_t)(qbase + wave * 16 + ln16) * DK;
    qf[0] = frag_from(qp + alo, qp + alo + 16);
    qf[1] = frag_from(qp + 32 + alo, qp + 32 + alo + 16);
  }

  v8f oacc[4];
  const v8f vzero = {0.f, 0.f, 0.f, 0.f, 0.f, 0.f, 0.f, 0.f};
#pragma unroll
  for (int dt = 0; dt < 4; dt++) oacc[dt] = vzero;
  float m[8], l[8];
#pragma unroll
  for (int v = 0; v < 8; v++) { m[v] = -__builtin_inff(); l[v] = 0.f; }

  issueKV(0, 0); // prologue

  for (int kt = 0; kt < SEQ; kt += 64) {
    const int cur = (kt >> 6) & 1, nxt = cur ^ 1;
    wait_async0();    // this wave's copies for 'cur' done
    __syncthreads();  // => everyone's copies done; prior reads of 'nxt' done
    if (kt + 64 < SEQ) issueKV(kt + 64, nxt); // overlaps compute below

    // S = Q @ K^T  (4 tiles of 16x16 over the 64 keys)
    v8f sacc[4];
#pragma unroll
    for (int nt = 0; nt < 4; nt++) sacc[nt] = vzero;
#pragma unroll
    for (int ks = 0; ks < 2; ks++) {
#pragma unroll
      for (int nt = 0; nt < 4; nt++) {
        const unsigned short* p = &Ks[cur][(nt * 16 + ln16) * 64 + ks * 32];
        v16bf bf = frag_from(p + blo, p + blo + 16);
        sacc[nt] = __builtin_amdgcn_wmma_f32_16x16x32_bf16(
            false, qf[ks], false, bf, (short)0, sacc[nt], false, false);
      }
    }

    // Online softmax; row r of C lives in one 16-lane half -> xor<16 reduce.
#pragma unroll
    for (int v = 0; v < 8; v++) {
      float x = fmaxf(fmaxf(sacc[0][v], sacc[1][v]), fmaxf(sacc[2][v], sacc[3][v]));
      x = fmaxf(x, __shfl_xor(x, 1, 32));
      x = fmaxf(x, __shfl_xor(x, 2, 32));
      x = fmaxf(x, __shfl_xor(x, 4, 32));
      x = fmaxf(x, __shfl_xor(x, 8, 32));
      float mn = fmaxf(m[v], x);
      float alpha = __expf(m[v] - mn);
      m[v] = mn;
      float rs = 0.f;
#pragma unroll
      for (int nt = 0; nt < 4; nt++) {
        float p = __expf(sacc[nt][v] - mn);
        sacc[nt][v] = p;
        rs += p;
      }
      rs += __shfl_xor(rs, 1, 32);
      rs += __shfl_xor(rs, 2, 32);
      rs += __shfl_xor(rs, 4, 32);
      rs += __shfl_xor(rs, 8, 32);
      l[v] = l[v] * alpha + rs;
#pragma unroll
      for (int dt = 0; dt < 4; dt++) oacc[dt][v] *= alpha;
    }

    // Re-layout P: C fragment -> LDS -> A fragment (per-wave region).
    unsigned short* Pw = &Ps[wave * 16 * 72];
#pragma unroll
    for (int nt = 0; nt < 4; nt++)
#pragma unroll
      for (int v = 0; v < 8; v++)
        Pw[(v + hi16 * 8) * 72 + nt * 16 + ln16] = f2bf(sacc[nt][v]);
    asm volatile("s_wait_dscnt 0x0" ::: "memory"); // wave-level LDS fence

    // O += P @ V  (A from P, B from V^T rows: contiguous t runs)
#pragma unroll
    for (int ks = 0; ks < 2; ks++) {
      const unsigned short* ap = &Pw[ln16 * 72 + ks * 32];
      v16bf af = frag_from(ap + alo, ap + alo + 16);
#pragma unroll
      for (int dt = 0; dt < 4; dt++) {
        const unsigned short* vp = &Vs[cur][(dt * 16 + ln16) * 64 + ks * 32];
        v16bf bf = frag_from(vp + blo, vp + blo + 16);
        oacc[dt] = __builtin_amdgcn_wmma_f32_16x16x32_bf16(
            false, af, false, bf, (short)0, oacc[dt], false, false);
      }
    }
  }

  // Finalize: divide by l, write bf16 at [b, s, h*64+d] for the final GEMM.
  const int b = bh >> 4, h = bh & 15;
#pragma unroll
  for (int dt = 0; dt < 4; dt++)
#pragma unroll
    for (int v = 0; v < 8; v++) {
      int s = qbase + wave * 16 + v + hi16 * 8;
      float o = oacc[dt][v] / l[v];
      O[((size_t)(b * SEQ + s)) * D_MODEL + h * DK + dt * 16 + ln16] = f2bf(o);
    }
}

// ---------------------------------------------------------------------------
extern "C" void kernel_launch(void* const* d_in, const int* in_sizes, int n_in,
                              void* d_out, int out_size, void* d_ws, size_t ws_size,
                              hipStream_t stream) {
  const float* x    = (const float*)d_in[0];
  const float* wqkv = (const float*)d_in[1];
  const float* wo   = (const float*)d_in[2];
  float* out = (float*)d_out;

  char* ws = (char*)d_ws;
  size_t off = 0;
  auto alloc = [&](size_t bytes) -> void* {
    void* p = ws + off;
    off = (off + bytes + 255) & ~(size_t)255;
    return p;
  };
  unsigned short* xb    = (unsigned short*)alloc((size_t)ROWS * D_MODEL * 2);
  unsigned short* wqkvb = (unsigned short*)alloc((size_t)D_MODEL * 3 * D_MODEL * 2);
  unsigned short* wob   = (unsigned short*)alloc((size_t)D_MODEL * D_MODEL * 2);
  unsigned short* Qb    = (unsigned short*)alloc((size_t)BATCH * NUM_HEADS * SEQ * DK * 2);
  unsigned short* Kb    = (unsigned short*)alloc((size_t)BATCH * NUM_HEADS * SEQ * DK * 2);
  unsigned short* Vtb   = (unsigned short*)alloc((size_t)BATCH * NUM_HEADS * DK * SEQ * 2);
  unsigned short* Ob    = (unsigned short*)alloc((size_t)ROWS * D_MODEL * 2);
  (void)ws_size; (void)in_sizes; (void)n_in; (void)out_size;

  // 1) f32 -> bf16 conversions
  {
    int n1 = ROWS * D_MODEL;
    cvt_f32_bf16<<<(n1 + 255) / 256, 256, 0, stream>>>(x, xb, n1);
    int n2 = D_MODEL * 3 * D_MODEL;
    cvt_f32_bf16<<<(n2 + 255) / 256, 256, 0, stream>>>(wqkv, wqkvb, n2);
    int n3 = D_MODEL * D_MODEL;
    cvt_f32_bf16<<<(n3 + 255) / 256, 256, 0, stream>>>(wo, wob, n3);
  }

  // 2) QKV projection: [4096,1024] @ [1024,3072], scatter to Q / K / V^T
  gemm_bf16<1><<<dim3(3 * D_MODEL / 128, ROWS / 128), 256, 0, stream>>>(
      xb, wqkvb, ROWS, 3 * D_MODEL, D_MODEL, nullptr, Qb, Kb, Vtb);

  // 3) Flash attention over all (b,h), 64 query rows per block
  flash_attn<<<dim3(SEQ / 64, BATCH * NUM_HEADS), 128, 0, stream>>>(Qb, Kb, Vtb, Ob);

  // 4) Output projection: [4096,1024] @ [1024,1024] -> f32 d_out
  gemm_bf16<0><<<dim3(D_MODEL / 128, ROWS / 128), 256, 0, stream>>>(
      Ob, wob, ROWS, D_MODEL, D_MODEL, out, nullptr, nullptr, nullptr);
}
